// RoPEMultiheadAttention_52097953300541
// MI455X (gfx1250) — matrix operand order repaired and checked by reading, not compiled, not run on hardware
//
#include <hip/hip_runtime.h>
#include <hip/hip_bf16.h>

// ---------------------------------------------------------------------------
// CDNA5 / gfx1250 wave32 WMMA bf16 implementation of RoPE-2D MHA.
// - All GEMM work on v_wmma_f32_16x16x32_bf16 (fp32 accumulate).
// - Flash attention K/V tiles are staged into LDS by the Tensor Data Mover
//   (tensor_load_to_lds, double-buffered, s_wait_tensorcnt + barrier).
// - Working set (~60MB) is L2-resident (192MB L2 @ 23.3 TB/s HBM behind it).
// ---------------------------------------------------------------------------

typedef __attribute__((ext_vector_type(16))) __bf16 v16bf;
typedef __attribute__((ext_vector_type(8)))  float  v8f;
typedef __attribute__((ext_vector_type(4)))  unsigned int v4u;
typedef __attribute__((ext_vector_type(4)))  unsigned int u32x4;
typedef __attribute__((ext_vector_type(8)))  int i32x8;
typedef __attribute__((ext_vector_type(4)))  int i32x4;

union Frag {
    v16bf          bf;
    v4u            q[2];   // two 16-byte chunks (ds_load_b128 / global_load_b128)
    unsigned short us[16];
};

__device__ __forceinline__ unsigned short f2bf(float f) {
    unsigned u = __float_as_uint(f);
    unsigned r = (u >> 16) & 1u;                       // round-to-nearest-even
    return (unsigned short)((u + 0x7fffu + r) >> 16);
}
__device__ __forceinline__ float bf2f(unsigned short h) {
    return __uint_as_float(((unsigned)h) << 16);
}
__device__ __forceinline__ v8f wmma_bf16(const Frag& a, const Frag& b, v8f c) {
    return __builtin_amdgcn_wmma_f32_16x16x32_bf16(
        /*neg_a=*/false, a.bf, /*neg_b=*/false, b.bf,
        /*c_mod=*/(short)0, c, /*reuse_a=*/false, /*reuse_b=*/false);
}

// ---------------------------------------------------------------------------
// Tensor DMA Descriptor (D#) packing per CDNA5 ISA.
// group0: [1:0]=count(1), [63:32]=lds_addr, [120:64]=global_addr, [127:126]=2
// group1: [17:16]=data_size(1 -> 2B), [20]=pad_enable, [24:22]=pad_interval,
//         [31:25]=pad_amount, [79:48]=tensor_dim0, [111:80]=tensor_dim1,
//         [127:112]=tile_dim0, [143:128]=tile_dim1, [159:144]=tile_dim2(0),
//         [207:160]=tensor_dim0_stride, [255:208]=tensor_dim1_stride(0)
// 2-D tensors need only groups 0-1; groups 2/3 are passed zeroed.
// ---------------------------------------------------------------------------
__device__ __forceinline__ u32x4 tdm_group0(unsigned lds_addr, const void* gptr) {
    unsigned long long ga = (unsigned long long)gptr;
    u32x4 g;
    g[0] = 1u;                                        // count=1, user mode
    g[1] = lds_addr;                                  // LDS byte address
    g[2] = (unsigned)(ga & 0xffffffffu);              // global_addr[31:0]
    g[3] = (unsigned)((ga >> 32) & 0x01ffffffu) | (2u << 30);  // [56:32] | type=2
    return g;
}
__device__ __forceinline__ i32x8 tdm_group1(unsigned dim0, unsigned dim1,
                                            unsigned long long stride0,
                                            unsigned tile0, unsigned tile1,
                                            unsigned pad_int, unsigned pad_amt) {
    i32x8 g;
    g[0] = (int)((1u << 16) | (1u << 20) | (pad_int << 22) | (pad_amt << 25));
    g[1] = (int)((dim0 & 0xffffu) << 16);             // tensor_dim0[15:0]
    g[2] = (int)((dim0 >> 16) | ((dim1 & 0xffffu) << 16));
    g[3] = (int)((dim1 >> 16) | (tile0 << 16));       // tile_dim0
    g[4] = (int)(tile1);                              // tile_dim1 (tile_dim2=0)
    g[5] = (int)(unsigned)(stride0 & 0xffffffffu);
    g[6] = (int)(unsigned)((stride0 >> 32) & 0xffffu);
    g[7] = 0;
    return g;
}

// ---------------------------------------------------------------------------
// Generic tiled GEMM: C[M,N] = A[M,K] * B[K,N], bf16 WMMA, fp32 accumulate.
// BM=BN=128, BK=32; 256 threads = 8 waves, each wave owns a 32x64 sub-tile.
// ---------------------------------------------------------------------------
template<bool A_BF16, bool OUT_BF16, bool BIAS>
__global__ void __launch_bounds__(256)
gemm_bf16(const void* __restrict__ Ap, const float* __restrict__ B,
          unsigned short* __restrict__ outBf, float* __restrict__ outF,
          const float* __restrict__ bias, int M, int N, int K) {
    __shared__ unsigned short sA[128][40];   // row-major, padded, 16B-aligned rows
    __shared__ unsigned short sB[128][40];   // column-major (B^T tile)

    const int tid  = threadIdx.x;
    const int lane = tid & 31;
    const int wave = tid >> 5;
    const int m0   = blockIdx.x * 128;
    const int n0   = blockIdx.y * 128;
    const int wr   = wave >> 1;
    const int wc   = wave & 1;
    const int cl    = lane & 15;
    const int lhalf = lane >> 4;

    const v8f zero8 = {0.f,0.f,0.f,0.f,0.f,0.f,0.f,0.f};
    v8f acc[2][4];
#pragma unroll
    for (int r = 0; r < 2; ++r)
#pragma unroll
        for (int c = 0; c < 4; ++c) acc[r][c] = zero8;

    for (int kk = 0; kk < K; kk += 32) {
        if constexpr (A_BF16) {
            const unsigned short* Ab = (const unsigned short*)Ap;
#pragma unroll
            for (int it = 0; it < 4; ++it) {
                int li = tid + it * 256;
                int r  = li >> 3, c4 = (li & 7) << 2;
                const unsigned short* src = Ab + (size_t)(m0 + r) * K + kk + c4;
                sA[r][c4+0] = src[0]; sA[r][c4+1] = src[1];
                sA[r][c4+2] = src[2]; sA[r][c4+3] = src[3];
            }
        } else {
            const float* Af = (const float*)Ap;
#pragma unroll
            for (int it = 0; it < 4; ++it) {
                int li = tid + it * 256;
                int r  = li >> 3, c4 = (li & 7) << 2;
                float4 v = *(const float4*)(Af + (size_t)(m0 + r) * K + kk + c4);
                sA[r][c4+0] = f2bf(v.x); sA[r][c4+1] = f2bf(v.y);
                sA[r][c4+2] = f2bf(v.z); sA[r][c4+3] = f2bf(v.w);
            }
        }
#pragma unroll
        for (int it = 0; it < 4; ++it) {
            int li = tid + it * 256;
            int kr = li >> 5, c4 = (li & 31) << 2;
            float4 v = *(const float4*)(B + (size_t)(kk + kr) * N + n0 + c4);
            sB[c4+0][kr] = f2bf(v.x); sB[c4+1][kr] = f2bf(v.y);
            sB[c4+2][kr] = f2bf(v.z); sB[c4+3][kr] = f2bf(v.w);
        }
        __syncthreads();

        Frag aF[2];
#pragma unroll
        for (int r = 0; r < 2; ++r) {
            int row = wr * 32 + r * 16 + cl;
            int k0  = lhalf * 8;
            aF[r].q[0] = *(const v4u*)&sA[row][k0];
            aF[r].q[1] = *(const v4u*)&sA[row][k0 + 16];
        }
#pragma unroll
        for (int c = 0; c < 4; ++c) {
            Frag bF;
            int col = wc * 64 + c * 16 + cl;
            int k0  = lhalf * 8;
            bF.q[0] = *(const v4u*)&sB[col][k0];
            bF.q[1] = *(const v4u*)&sB[col][k0 + 16];
#pragma unroll
            for (int r = 0; r < 2; ++r)
                acc[r][c] = wmma_bf16(aF[r], bF, acc[r][c]);
        }
        __syncthreads();
    }

#pragma unroll
    for (int r = 0; r < 2; ++r)
#pragma unroll
        for (int c = 0; c < 4; ++c) {
            int gr0 = m0 + wr * 32 + r * 16 + 8 * lhalf;
            int gc  = n0 + wc * 64 + c * 16 + cl;
            float bv = BIAS ? bias[gc] : 0.f;
#pragma unroll
            for (int v = 0; v < 8; ++v) {
                if constexpr (OUT_BF16)
                    outBf[(size_t)(gr0 + v) * N + gc] = f2bf(acc[r][c][v]);
                else
                    outF[(size_t)(gr0 + v) * N + gc] = acc[r][c][v] + bv;
            }
        }
}

// ---------------------------------------------------------------------------
// RoPE-2D + head split. qkv[N*T,1536] (bf16) ->
//   qR,kR : [N,H,T,64] rotated bf16 ; vT : [N,H,64,T] transposed bf16
// ---------------------------------------------------------------------------
__global__ void __launch_bounds__(256)
rope_split(const unsigned short* __restrict__ qkv,
           const int* __restrict__ pos_h, const int* __restrict__ pos_w,
           unsigned short* __restrict__ qR, unsigned short* __restrict__ kR,
           unsigned short* __restrict__ vT) {
    int idx = blockIdx.x * 256 + threadIdx.x;
    int p  = idx & 31;
    int h  = (idx >> 5) & 7;
    int t  = (idx >> 8) & 2047;
    int n  = idx >> 19;
    int hf = p >> 4;
    int i  = p & 15;

    int   pos  = hf ? pos_w[t] : pos_h[t];
    float freq = __expf(-(float)i * 0.57564627324851149f);   // theta^(-i/16)
    float ang  = (float)pos * freq;
    float s, c;
    __sincosf(ang, &s, &c);

    int    dd  = hf * 32 + 2 * i;
    size_t row = (size_t)(n * 2048 + t) * 1536;
    int    qc  = h * 64 + dd;
    size_t ho  = (size_t)((n * 8 + h) * 2048 + t) * 64 + dd;

    float x1 = bf2f(qkv[row + qc]),       x2 = bf2f(qkv[row + qc + 1]);
    qR[ho]     = f2bf(x1 * c - x2 * s);
    qR[ho + 1] = f2bf(x1 * s + x2 * c);

    float y1 = bf2f(qkv[row + 512 + qc]), y2 = bf2f(qkv[row + 512 + qc + 1]);
    kR[ho]     = f2bf(y1 * c - y2 * s);
    kR[ho + 1] = f2bf(y1 * s + y2 * c);

    size_t vo = (size_t)((n * 8 + h) * 64 + dd) * 2048 + t;
    vT[vo]        = qkv[row + 1024 + qc];
    vT[vo + 2048] = qkv[row + 1024 + qc + 1];
}

// ---------------------------------------------------------------------------
// Flash attention. One (n,h) per blockIdx.y, 64 query rows per block,
// 4 waves x 16 rows. K/V tiles staged by TDM into double-buffered LDS:
//   K tile: 32 tokens x 64 dims, contiguous 4KB, TDM pads 4 DW per 32-DW row
//           -> LDS stride 36 DW (72 bf16), bank-conflict-friendly b128 reads.
//   V tile: 64 dims x 32 tokens (row stride 2048), pads 4 DW per 16-DW row
//           -> LDS stride 20 DW (40 bf16).
// Wave 0 drives the TDM; s_wait_tensorcnt + barrier publishes each buffer.
// ---------------------------------------------------------------------------
__global__ void __launch_bounds__(128)
flash_attn(const unsigned short* __restrict__ Q,
           const unsigned short* __restrict__ Kh,
           const unsigned short* __restrict__ Vh,
           unsigned short* __restrict__ O) {
    __shared__ unsigned short sK[2][32][72];   // 9.2KB, TDM-padded K tiles
    __shared__ unsigned short sV[2][64][40];   // 10.2KB, TDM-padded V^T tiles
    __shared__ unsigned short sP[4][16][32];   // per-wave P relayout buffer

    const int tid   = threadIdx.x;
    const int lane  = tid & 31;
    const int wave  = tid >> 5;
    const int cl    = lane & 15;
    const int lhalf = lane >> 4;
    const int bh    = blockIdx.y;              // n*8 + h
    const int n     = bh >> 3, h = bh & 7;

    const unsigned short* Qb = Q  + (size_t)bh * 2048 * 64;
    const unsigned short* Kb = Kh + (size_t)bh * 2048 * 64;
    const unsigned short* Vb = Vh + (size_t)bh * 64 * 2048;
    const int q0 = blockIdx.x * 64 + wave * 16;

    // Loop-invariant D# group1 descriptors (wave-uniform -> SGPRs).
    const i32x8 g1K = tdm_group1(/*dim0*/2048u * 64u, /*dim1*/1u,
                                 /*stride0*/2048ull, /*tile0*/2048u, /*tile1*/0u,
                                 /*pad_int(32DW)*/4u, /*pad_amt(4DW)*/3u);
    const i32x8 g1V = tdm_group1(/*dim0*/2048u, /*dim1*/64u,
                                 /*stride0*/2048ull, /*tile0*/32u, /*tile1*/64u,
                                 /*pad_int(16DW)*/3u, /*pad_amt(4DW)*/3u);
    const i32x4 z4 = {0, 0, 0, 0};
    const i32x8 z8 = {0, 0, 0, 0, 0, 0, 0, 0};

    // Q A-fragments (16 rows x 64 dims = 2 k-steps), kept in registers.
    Frag qf[2];
#pragma unroll
    for (int ks = 0; ks < 2; ++ks) {
        const unsigned short* src = Qb + (size_t)(q0 + cl) * 64 + ks * 32 + lhalf * 8;
        qf[ks].q[0] = *(const v4u*)src;
        qf[ks].q[1] = *(const v4u*)(src + 16);
    }

    const v8f zero8 = {0.f,0.f,0.f,0.f,0.f,0.f,0.f,0.f};
    float m[8], l[8];
    v8f   oacc[4];
#pragma unroll
    for (int v = 0; v < 8; ++v) { m[v] = -1e30f; l[v] = 0.f; }
#pragma unroll
    for (int c = 0; c < 4; ++c) oacc[c] = zero8;

    // Prologue: TDM-load tile 0 into buffer 0.
    if (wave == 0) {
        __builtin_amdgcn_tensor_load_to_lds(
            tdm_group0((unsigned)(size_t)&sK[0][0][0], Kb), g1K, z4, z4, z8, 0);
        __builtin_amdgcn_tensor_load_to_lds(
            tdm_group0((unsigned)(size_t)&sV[0][0][0], Vb), g1V, z4, z4, z8, 0);
        __builtin_amdgcn_s_wait_tensorcnt(0);
    }
    __syncthreads();

    for (int kb = 0; kb < 2048; kb += 32) {
        const int  cur      = (kb >> 5) & 1;
        const int  nxt      = cur ^ 1;
        const bool has_next = (kb + 32) < 2048;

        // Kick off the next tile while this one is consumed.
        if (wave == 0 && has_next) {
            __builtin_amdgcn_tensor_load_to_lds(
                tdm_group0((unsigned)(size_t)&sK[nxt][0][0],
                           Kb + (size_t)(kb + 32) * 64), g1K, z4, z4, z8, 0);
            __builtin_amdgcn_tensor_load_to_lds(
                tdm_group0((unsigned)(size_t)&sV[nxt][0][0],
                           Vb + (kb + 32)), g1V, z4, z4, z8, 0);
        }

        // ---- S = Q K^T for 32 keys (2 key sub-tiles x 2 k-steps) ----------
        v8f s[2];
#pragma unroll
        for (int j = 0; j < 2; ++j) {
            v8f sj = zero8;
#pragma unroll
            for (int ks = 0; ks < 2; ++ks) {
                Frag kf;
                const unsigned short* src = &sK[cur][j * 16 + cl][ks * 32 + lhalf * 8];
                kf.q[0] = *(const v4u*)src;
                kf.q[1] = *(const v4u*)(src + 16);
                sj = wmma_bf16(qf[ks], kf, sj);
            }
            s[j] = sj;
        }
        // ---- online softmax; row stats via half-wave shfl reductions ------
#pragma unroll
        for (int v = 0; v < 8; ++v) {
            float a = s[0][v] * 0.125f;        // 1/sqrt(64)
            float b = s[1][v] * 0.125f;
            float tmax = fmaxf(a, b);
#pragma unroll
            for (int off = 1; off < 16; off <<= 1)
                tmax = fmaxf(tmax, __shfl_xor(tmax, off, 32));
            float mn    = fmaxf(m[v], tmax);
            float alpha = __expf(m[v] - mn);
            m[v] = mn;
            float p0 = __expf(a - mn);
            float p1 = __expf(b - mn);
            float rs = p0 + p1;
#pragma unroll
            for (int off = 1; off < 16; off <<= 1)
                rs += __shfl_xor(rs, off, 32);
            l[v] = l[v] * alpha + rs;
#pragma unroll
            for (int c = 0; c < 4; ++c) oacc[c][v] *= alpha;
            sP[wave][8 * lhalf + v][cl]      = f2bf(p0);
            sP[wave][8 * lhalf + v][cl + 16] = f2bf(p1);
        }
        // ---- P as A-fragment, O += P V ------------------------------------
        Frag pf;
        pf.q[0] = *(const v4u*)&sP[wave][cl][lhalf * 8];
        pf.q[1] = *(const v4u*)&sP[wave][cl][lhalf * 8 + 16];
#pragma unroll
        for (int c = 0; c < 4; ++c) {
            Frag vf;
            const unsigned short* src = &sV[cur][c * 16 + cl][lhalf * 8];
            vf.q[0] = *(const v4u*)src;
            vf.q[1] = *(const v4u*)(src + 16);
            oacc[c] = wmma_bf16(pf, vf, oacc[c]);
        }

        // Publish buffer nxt: TDM done (wave 0) + all waves done reading cur.
        if (wave == 0 && has_next)
            __builtin_amdgcn_s_wait_tensorcnt(0);
        __syncthreads();
    }

    // ---- O /= l, write bf16 to [N*T, 512] token-major ---------------------
    unsigned short* Ob = O + (size_t)n * 2048 * 512 + (size_t)h * 64;
#pragma unroll
    for (int v = 0; v < 8; ++v) {
        float inv = 1.0f / l[v];
        int row = q0 + 8 * lhalf + v;
#pragma unroll
        for (int c = 0; c < 4; ++c)
            Ob[(size_t)row * 512 + c * 16 + cl] = f2bf(oacc[c][v] * inv);
    }
}

// ---------------------------------------------------------------------------
extern "C" void kernel_launch(void* const* d_in, const int* in_sizes, int n_in,
                              void* d_out, int out_size, void* d_ws, size_t ws_size,
                              hipStream_t stream) {
    const float* x      = (const float*)d_in[0];
    const int*   pos_h  = (const int*)d_in[1];
    const int*   pos_w  = (const int*)d_in[2];
    const float* w_qkv  = (const float*)d_in[3];
    const float* w_proj = (const float*)d_in[4];
    const float* b_proj = (const float*)d_in[5];
    float*       out    = (float*)d_out;

    // Workspace layout (bf16 intermediates), ~58.7 MB total:
    char* ws = (char*)d_ws;
    unsigned short* qkv = (unsigned short*)(ws);              // 8192*1536*2 = 25.2MB
    unsigned short* qR  = (unsigned short*)(ws + 25165824);   // 8.4MB
    unsigned short* kR  = (unsigned short*)(ws + 33554432);   // 8.4MB
    unsigned short* vT  = (unsigned short*)(ws + 41943040);   // 8.4MB
    unsigned short* oB  = (unsigned short*)(ws + 50331648);   // 8.4MB

    // 1) qkv = x @ w_qkv        (8192 x 1536 x 512, bf16 out)
    gemm_bf16<false, true, false><<<dim3(64, 12), 256, 0, stream>>>(
        x, w_qkv, qkv, nullptr, nullptr, 8192, 1536, 512);
    // 2) RoPE-2D + head split + V transpose
    rope_split<<<8192, 256, 0, stream>>>(qkv, pos_h, pos_w, qR, kR, vT);
    // 3) flash attention (32 q-tiles x 32 (n,h) pairs), TDM-fed
    flash_attn<<<dim3(32, 32), 128, 0, stream>>>(qR, kR, vT, oB);
    // 4) out = o @ w_proj + b   (8192 x 512 x 512, fp32 out)
    gemm_bf16<true, false, true><<<dim3(64, 4), 256, 0, stream>>>(
        oB, w_proj, nullptr, out, b_proj, 8192, 512, 512);
}
